// QwenSparseMoeBlock_60275571032328
// MI455X (gfx1250) — matrix operand order repaired and checked by reading, not compiled, not run on hardware
//
#include <hip/hip_runtime.h>

typedef __attribute__((ext_vector_type(16))) _Float16 v16h;
typedef __attribute__((ext_vector_type(8)))  _Float16 v8h;
typedef __attribute__((ext_vector_type(4)))  _Float16 v4h;
typedef __attribute__((ext_vector_type(2)))  _Float16 v2h;
typedef __attribute__((ext_vector_type(8)))  float    v8f;

static constexpr int kH  = 2048;   // hidden
static constexpr int kE  = 8;      // experts
static constexpr int kI  = 1408;   // expert intermediate
static constexpr int kIS = 5632;   // shared intermediate
static constexpr int kT  = 1024;   // tokens (2*512)

static constexpr int BM = 128, BN = 64, BK = 32;

#define ASYNC_WAIT() asm volatile("s_wait_asynccnt 0x0" ::: "memory")

// ---------------------------------------------------------------- utilities

__global__ __launch_bounds__(256) void zero_f4(float4* __restrict__ p) {
  p[(size_t)blockIdx.x * 256 + threadIdx.x] = make_float4(0.f, 0.f, 0.f, 0.f);
}

__global__ __launch_bounds__(256) void cvt_f32_f16(const float4* __restrict__ s,
                                                   v4h* __restrict__ d) {
  size_t i = (size_t)blockIdx.x * 256 + threadIdx.x;
  float4 v = s[i];
  v4h h;
  h[0] = (_Float16)v.x; h[1] = (_Float16)v.y;
  h[2] = (_Float16)v.z; h[3] = (_Float16)v.w;
  d[i] = h;
}

// ---------------------------------------------------------------- router
__global__ __launch_bounds__(256)
void router_kernel(const float* __restrict__ x, const float* __restrict__ gw,
                   const float* __restrict__ sgw, float* __restrict__ logits_out,
                   float* __restrict__ combine, float* __restrict__ sgate) {
  __shared__ float red[256];
  __shared__ float sums[9];
  const int t = blockIdx.x, tid = threadIdx.x;
  float part[9];
#pragma unroll
  for (int v = 0; v < 9; ++v) part[v] = 0.f;
#pragma unroll
  for (int k = 0; k < kH / 256; ++k) {
    const int h = tid + k * 256;
    const float xv = x[(size_t)t * kH + h];
#pragma unroll
    for (int e = 0; e < kE; ++e) part[e] += xv * gw[(size_t)h * kE + e];
    part[8] += xv * sgw[h];
  }
  for (int v = 0; v < 9; ++v) {
    red[tid] = part[v];
    __syncthreads();
    for (int s = 128; s > 0; s >>= 1) {
      if (tid < s) red[tid] += red[tid + s];
      __syncthreads();
    }
    if (tid == 0) sums[v] = red[0];
    __syncthreads();
  }
  if (tid == 0) {
    float lg[kE], mx = -3.0e38f;
#pragma unroll
    for (int e = 0; e < kE; ++e) {
      lg[e] = sums[e];
      logits_out[(size_t)t * kE + e] = lg[e];
      mx = fmaxf(mx, lg[e]);
    }
    float p[kE], s = 0.f;
#pragma unroll
    for (int e = 0; e < kE; ++e) { p[e] = __expf(lg[e] - mx); s += p[e]; }
    const float inv = 1.f / s;
#pragma unroll
    for (int e = 0; e < kE; ++e) p[e] *= inv;
    int e1 = 0;
    for (int e = 1; e < kE; ++e) if (p[e] > p[e1]) e1 = e;
    int e2 = (e1 == 0) ? 1 : 0;
    for (int e = 0; e < kE; ++e) if (e != e1 && p[e] > p[e2]) e2 = e;
#pragma unroll
    for (int e = 0; e < kE; ++e)
      combine[(size_t)t * kE + e] = (e == e1) ? p[e1] : ((e == e2) ? p[e2] : 0.f);
    sgate[t] = 1.f / (1.f + __expf(-sums[8]));
  }
}

// ---------------------------------------------------------------- WMMA GEMM
// MODE 0: C_half = silu(A*B0) * (A*B1)      (gate/up fused, fp16 out)
// MODE 1: C_f32[m,n] += scale[m]*(A*B0)     (down proj, scaled accumulate)
// A: [M,K] fp16 row-major (staged via async global->LDS, ASYNCcnt-tracked).
// B: [K,N] fp32 row-major, fp32->fp16 converted in-register, packed b32 LDS
//    stores into the WMMA fragment-swizzled layout. Double-buffered LDS:
//    one barrier per k-iteration.
template <int MODE>
__global__ __launch_bounds__(256)
void moe_gemm(const _Float16* __restrict__ A, int lda,
              const float* __restrict__ B0, const float* __restrict__ B1, int ldb,
              void* __restrict__ Cv, int ldc,
              const float* __restrict__ scale, int sstride, int K) {
  constexpr int MATS = (MODE == 0) ? 2 : 1;
  __shared__ alignas(16) _Float16 As[2][BM * BK];        // fragment-swizzled
  __shared__ alignas(16) _Float16 Bs[MATS][2][BN * BK];  // fragment-swizzled

  const int tid  = threadIdx.x;
  const int lane = tid & 31;
  const int wv   = tid >> 5;   // 0..7
  const int wm   = wv >> 1;    // 0..3  (row group of 32)
  const int wn   = wv & 1;     // 0..1  (col group of 32)
  const int m0   = blockIdx.x * BM;
  const int n0   = blockIdx.y * BN;
  const int KT   = K / BK;

  float bpre[MATS][8];

  // A tile: 512 x 16B chunks; each thread issues 2 async global->LDS b128,
  // destination already in the per-lane WMMA fragment layout.
  auto g_asyncA = [&](int kt, int buf) {
#pragma unroll
    for (int it = 0; it < 2; ++it) {
      const int c = tid + it * 256;
      const int m = c >> 2, kc = (c & 3) * 8;
      const unsigned long long ga =
          (unsigned long long)(A + (size_t)(m0 + m) * lda + kt * BK + kc);
      const int lp = (m & 15) | (((kc >> 3) & 1) << 4);
      const int i0 = ((kc >> 4) & 1) * 8;
      const unsigned int lo =
          (unsigned int)(uintptr_t)&As[buf][(m >> 4) * 512 + lp * 16 + i0];
      asm volatile("global_load_async_to_lds_b128 %0, %1, off"
                   :: "v"(lo), "v"(ga) : "memory");
    }
  };
  // B tile: k-pairs so two converted halves pack into one b32 LDS store.
  auto g_loadB = [&](int kt) {
#pragma unroll
    for (int j = 0; j < 4; ++j) {
      const int idx = tid + j * 256;          // pair index
      const int n = idx & 63, kp = idx >> 6;  // kp = 0..15
      const size_t off = (size_t)(kt * BK + kp * 2) * ldb + n0 + n;
      bpre[0][2 * j]     = B0[off];
      bpre[0][2 * j + 1] = B0[off + ldb];
      if (MATS == 2) {
        bpre[1][2 * j]     = B1[off];
        bpre[1][2 * j + 1] = B1[off + ldb];
      }
      if (kt + 2 < KT)  // stream-prefetch weights 2 tiles ahead
        __builtin_prefetch(B0 + off + (size_t)2 * BK * ldb, 0, 1);
    }
  };
  auto s_storeB = [&](int buf) {
#pragma unroll
    for (int j = 0; j < 4; ++j) {
      const int idx = tid + j * 256;
      const int n = idx & 63, k0 = (idx >> 6) * 2;
      const int lp  = (n & 15) | (((k0 >> 3) & 1) << 4);
      const int off = (n >> 4) * 512 + lp * 16 + (k0 & 7) + ((k0 >> 4) & 1) * 8;
#pragma unroll
      for (int mt = 0; mt < MATS; ++mt) {
        v2h p;
        p[0] = (_Float16)bpre[mt][2 * j];
        p[1] = (_Float16)bpre[mt][2 * j + 1];
        *(v2h*)&Bs[mt][buf][off] = p;   // one ds_store_b32
      }
    }
  };

  v8f acc[MATS][2][2];
#pragma unroll
  for (int a0 = 0; a0 < MATS; ++a0)
#pragma unroll
    for (int i = 0; i < 2; ++i)
#pragma unroll
      for (int j = 0; j < 2; ++j)
#pragma unroll
        for (int q = 0; q < 8; ++q) acc[a0][i][j][q] = 0.f;

  // prologue: fill buffer 0
  g_asyncA(0, 0);
  g_loadB(0);
  s_storeB(0);
  ASYNC_WAIT();
  __syncthreads();

  for (int kt = 0; kt < KT; ++kt) {
    const int cur = kt & 1, nxt = cur ^ 1;
    const bool more = (kt + 1 < KT);
    if (more) {                    // next tile: async A + B regs, overlapped
      g_asyncA(kt + 1, nxt);
      g_loadB(kt + 1);
    }
    v16h af[2];
#pragma unroll
    for (int im = 0; im < 2; ++im)
      af[im] = *(const v16h*)(As[cur] + (wm * 2 + im) * 512 + lane * 16);
#pragma unroll
    for (int mt = 0; mt < MATS; ++mt)
#pragma unroll
      for (int jn = 0; jn < 2; ++jn) {
        const v16h bf =
            *(const v16h*)(Bs[mt][cur] + (wn * 2 + jn) * 512 + lane * 16);
#pragma unroll
        for (int im = 0; im < 2; ++im)
          acc[mt][im][jn] = __builtin_amdgcn_wmma_f32_16x16x32_f16(
              false, af[im], false, bf, (short)0, acc[mt][im][jn], false, false);
      }
    if (more) s_storeB(nxt);       // nxt not read until after the barrier
    ASYNC_WAIT();                  // As[nxt] writes landed
    __syncthreads();               // single barrier per k-iteration
  }

  // C/D layout: vgpr j -> M = j + (lane>=16)*8 ; N = lane&15
  const int mb = m0 + wm * 32 + ((lane >> 4) << 3);
  const int nb = n0 + wn * 32 + (lane & 15);
  if (MODE == 0) {
    _Float16* C = (_Float16*)Cv;
#pragma unroll
    for (int im = 0; im < 2; ++im)
#pragma unroll
      for (int jn = 0; jn < 2; ++jn)
#pragma unroll
        for (int j = 0; j < 8; ++j) {
          const float g = acc[0][im][jn][j];
          const float u = acc[1][im][jn][j];
          const float h = (g / (1.f + __expf(-g))) * u;   // silu(g)*u
          C[(size_t)(mb + im * 16 + j) * ldc + nb + jn * 16] = (_Float16)h;
        }
  } else {
    float* C = (float*)Cv;
#pragma unroll
    for (int im = 0; im < 2; ++im)
#pragma unroll
      for (int jn = 0; jn < 2; ++jn)
#pragma unroll
        for (int j = 0; j < 8; ++j) {
          const int m = mb + im * 16 + j;
          const float sc = scale[(size_t)m * sstride];
          C[(size_t)m * ldc + nb + jn * 16] += sc * acc[0][im][jn][j];
        }
  }
}

// ---------------------------------------------------------------- launch

extern "C" void kernel_launch(void* const* d_in, const int* in_sizes, int n_in,
                              void* d_out, int out_size, void* d_ws, size_t ws_size,
                              hipStream_t stream) {
  (void)in_sizes; (void)n_in; (void)out_size; (void)ws_size;
  const float* x   = (const float*)d_in[0];  // [2,512,H]
  const float* gw  = (const float*)d_in[1];  // [H,E]
  const float* wg  = (const float*)d_in[2];  // [E,H,I]
  const float* wu  = (const float*)d_in[3];  // [E,H,I]
  const float* wd  = (const float*)d_in[4];  // [E,I,H]
  const float* sg  = (const float*)d_in[5];  // [H,IS]
  const float* su  = (const float*)d_in[6];  // [H,IS]
  const float* sd  = (const float*)d_in[7];  // [IS,H]
  const float* sgw = (const float*)d_in[8];  // [H,1]

  float* out    = (float*)d_out;             // final [T,H]
  float* logits = out + (size_t)kT * kH;     // router_logits [T,E]

  char* ws = (char*)d_ws;
  _Float16* Xf      = (_Float16*)ws;                     // T*H fp16
  size_t    o       = (size_t)kT * kH * 2;
  _Float16* Hbuf    = (_Float16*)(ws + o);               // T*IS fp16 (reused)
  o += (size_t)kT * kIS * 2;
  float*    combine = (float*)(ws + o);                  // T*E
  o += (size_t)kT * kE * 4;
  float*    sgate   = (float*)(ws + o);                  // T

  const dim3 blk(256);

  zero_f4<<<(kT * kH / 4) / 256, blk, 0, stream>>>((float4*)out);
  cvt_f32_f16<<<(kT * kH / 4) / 256, blk, 0, stream>>>((const float4*)x, (v4h*)Xf);
  router_kernel<<<kT, blk, 0, stream>>>(x, gw, sgw, logits, combine, sgate);

  for (int e = 0; e < kE; ++e) {
    const size_t wo   = (size_t)e * kH * kI;
    const size_t doff = (size_t)e * kI * kH;
    moe_gemm<0><<<dim3(kT / BM, kI / BN), blk, 0, stream>>>(
        Xf, kH, wg + wo, wu + wo, kI, Hbuf, kI, nullptr, 0, kH);
    moe_gemm<1><<<dim3(kT / BM, kH / BN), blk, 0, stream>>>(
        Hbuf, kI, wd + doff, nullptr, kH, out, kH, combine + e, kE, kI);
  }
  moe_gemm<0><<<dim3(kT / BM, kIS / BN), blk, 0, stream>>>(
      Xf, kH, sg, su, kIS, Hbuf, kIS, nullptr, 0, kH);
  moe_gemm<1><<<dim3(kT / BM, kH / BN), blk, 0, stream>>>(
      Hbuf, kIS, sd, nullptr, kH, out, kH, sgate, 1, kIS);
}
